// SmolLM_78202764525984
// MI455X (gfx1250) — compile-verified, matrix operand order checked by reading, not run on hardware
//
#include <hip/hip_runtime.h>
#include <hip/hip_fp16.h>

// ---------------------------------------------------------------------------
// SmolLM-ish GPT forward for gfx1250 (MI455X).
// - All GEMMs on v_wmma_f32_16x16x32_f16 (f16 in, f32 accumulate)
// - Register double-buffered GEMM inner loop (overlap loads with WMMA)
// - Attention K/V tiles staged to LDS via the Tensor Data Mover
//   (tensor_load_to_lds + s_wait_tensorcnt), with portable fallback.
// - Fragment loads hoisted ahead of WMMA issue so waits are batched.
// ---------------------------------------------------------------------------

typedef __attribute__((ext_vector_type(16))) _Float16 v16h;
typedef __attribute__((ext_vector_type(8)))  float    v8f;

#define C_B       4
#define C_T       1024
#define C_D       576
#define C_NL      30
#define C_NH      9
#define C_NKV     3
#define C_HD      64
#define C_MLP     1536
#define C_V       49152
#define C_M       (C_B * C_T)      // 4096 rows
#define C_KVW     (C_NKV * C_HD)   // 192

#if defined(__HIP_DEVICE_COMPILE__) && __has_builtin(__builtin_amdgcn_tensor_load_to_lds)
#define HAS_TDM 1
#else
#define HAS_TDM 0
#endif

#if defined(__HIP_DEVICE_COMPILE__) && !HAS_TDM
#warning "CDNA5 probe: __builtin_amdgcn_tensor_load_to_lds NOT available; using plain LDS staging fallback"
#endif

// ---------------------------------------------------------------------------
// device helpers
// ---------------------------------------------------------------------------

__device__ __forceinline__ v8f zero8() {
  v8f z;
#pragma unroll
  for (int i = 0; i < 8; ++i) z[i] = 0.0f;
  return z;
}

// Load a 16x32 f16 fragment (A-layout / Bt-layout) from row-major memory with
// leading dimension ld (in halfs). Lane l<16 holds row l, K={0..7,16..23};
// lane l>=16 holds row l-16, K={8..15,24..31}.  (ISA 7.12.2, 16-bit A 16x32)
__device__ __forceinline__ v16h load_frag(const _Float16* base, int ld) {
  const int lane = (int)(threadIdx.x & 31u);
  const int r    = lane & 15;
  const int kh   = (lane >> 4) << 3;            // 0 or 8
  const _Float16* p = base + (size_t)r * ld + kh;
  union { v16h v; uint4 u[2]; } t;
  t.u[0] = *reinterpret_cast<const uint4*>(p);        // K = kh .. kh+7
  t.u[1] = *reinterpret_cast<const uint4*>(p + 16);   // K = kh+16 .. kh+23
  return t.v;
}

__device__ __forceinline__ v8f wmma32(v16h a, v16h b, v8f c) {
  return __builtin_amdgcn_wmma_f32_16x16x32_f16(
      /*neg_a=*/false, a, /*neg_b=*/false, b,
      /*c_mod=*/(short)0, c, /*reuse_a=*/false, /*reuse_b=*/false);
}

#if HAS_TDM
// TDM: async DMA of a 32-row x 64-element (f16) 2D tile, rows strided by
// rowStrideElems halfs, from global memory into contiguous LDS.
// Descriptor per CDNA5 ISA ch.8: group0 = {count,lds_addr,global_addr,type=2},
// group1 = {data_size=2B, tensor dims, tile dims 64x32, dim0 stride}.
__device__ __forceinline__ void tdm_load_tile32x64(unsigned ldsByteAddr,
                                                   const _Float16* gsrc,
                                                   int rowStrideElems) {
  typedef unsigned int u32x4 __attribute__((ext_vector_type(4)));
  typedef int i32x4 __attribute__((ext_vector_type(4)));
  typedef int i32x8 __attribute__((ext_vector_type(8)));
  unsigned long long ga = (unsigned long long)(uintptr_t)gsrc;
  u32x4 g0;
  g0[0] = 1u;                                              // count=1 (user D#)
  g0[1] = ldsByteAddr;                                     // lds_addr
  g0[2] = (unsigned)(ga & 0xffffffffu);                    // global_addr[31:0]
  g0[3] = (unsigned)((ga >> 32) & 0x1ffffffu) | (2u << 30);// addr[56:32]|type=2
  union { unsigned u[8]; i32x8 v; } g1;
  g1.u[0] = (1u << 16);                          // wg_mask=0, data_size=2B
  g1.u[1] = ((unsigned)rowStrideElems) << 16;    // tensor_dim0 (low 16)
  g1.u[2] = ((unsigned)C_T) << 16;               // tensor_dim1 (low 16)
  g1.u[3] = (64u << 16);                         // tile_dim0 = 64
  g1.u[4] = 32u;                                 // tile_dim1 = 32, tile_dim2=0
  g1.u[5] = (unsigned)rowStrideElems;            // tensor_dim0_stride[31:0]
  g1.u[6] = 0u;
  g1.u[7] = 0u;
  i32x4 z4 = 0;
#if __clang_major__ >= 23
  i32x8 z8 = 0;
  __builtin_amdgcn_tensor_load_to_lds(g0, g1.v, z4, z4, z8, 0);
#else
  __builtin_amdgcn_tensor_load_to_lds(g0, g1.v, z4, z4, 0);
#endif
}
#endif

// ---------------------------------------------------------------------------
// weight conversion: fp32 [K,N] -> f16 [N,K]   (so B-fragments load like A)
// ---------------------------------------------------------------------------
__global__ void transpose_f16_kernel(const float* __restrict__ W,
                                     _Float16* __restrict__ Wt, int K, int N) {
  int idx = blockIdx.x * blockDim.x + threadIdx.x;   // idx = n*K + k
  int total = K * N;
  if (idx < total) {
    int n = idx / K;
    int k = idx - n * K;
    Wt[idx] = (_Float16)W[(size_t)k * N + n];
  }
}

__global__ void cast_f16_kernel(const float* __restrict__ W,
                                _Float16* __restrict__ Wt, long long n) {
  long long stride = (long long)gridDim.x * blockDim.x;
  for (long long i = (long long)blockIdx.x * blockDim.x + threadIdx.x; i < n; i += stride)
    Wt[i] = (_Float16)W[i];
}

// ---------------------------------------------------------------------------
// embedding: x[bt, :] = wte[idx[bt], :] + wpe[t, :]
// ---------------------------------------------------------------------------
__global__ void embed_kernel(const int* __restrict__ idx,
                             const float* __restrict__ wte,
                             const float* __restrict__ wpe,
                             float* __restrict__ x) {
  int bt  = blockIdx.x;
  int t   = bt & (C_T - 1);
  int tok = idx[bt];
  for (int i = threadIdx.x; i < C_D; i += blockDim.x)
    x[(size_t)bt * C_D + i] = wte[(size_t)tok * C_D + i] + wpe[(size_t)t * C_D + i];
}

// ---------------------------------------------------------------------------
// layernorm over D=576, fp32 in -> f16 out (feeds WMMA GEMMs)
// ---------------------------------------------------------------------------
__global__ __launch_bounds__(256)
void layernorm_f16_kernel(const float* __restrict__ x,
                          const float* __restrict__ w,
                          const float* __restrict__ b,
                          _Float16* __restrict__ out) {
  const int row  = blockIdx.x;
  const int lane = threadIdx.x & 31;
  const int wv   = threadIdx.x >> 5;
  const float* xr = x + (size_t)row * C_D;

  float s = 0.0f, s2 = 0.0f;
  for (int i = threadIdx.x; i < C_D; i += 256) {
    float v = xr[i];
    s += v; s2 += v * v;
  }
#pragma unroll
  for (int o = 16; o >= 1; o >>= 1) {
    s  += __shfl_xor(s,  o, 32);
    s2 += __shfl_xor(s2, o, 32);
  }
  __shared__ float red[16];
  __shared__ float mu_s, rs_s;
  if (lane == 0) { red[wv] = s; red[8 + wv] = s2; }
  __syncthreads();
  if (threadIdx.x == 0) {
    float a = 0.0f, c = 0.0f;
#pragma unroll
    for (int j = 0; j < 8; ++j) { a += red[j]; c += red[8 + j]; }
    float mu  = a / (float)C_D;
    float var = c / (float)C_D - mu * mu;
    mu_s = mu;
    rs_s = rsqrtf(var + 1e-5f);
  }
  __syncthreads();
  float mu = mu_s, rs = rs_s;
  for (int i = threadIdx.x; i < C_D; i += 256)
    out[(size_t)row * C_D + i] = (_Float16)((xr[i] - mu) * rs * w[i] + b[i]);
}

// ---------------------------------------------------------------------------
// WMMA GEMM:  C[M,N] = A[M,K] (f16, row-major) @ Bt[N,K]^T (f16) [+ bias][+res]
// block = 256 threads = 8 waves, tile 128(M) x 64(N), each wave 32x32.
// Register double-buffered K loop: next fragments load while current WMMAs run.
// MODE 0: f16 out = acc + bias           (q,k,v)
// MODE 1: f16 out = silu(acc + bias)     (mlp fc)
// MODE 2: f32 out = acc + bias + resid   (attn out-proj, mlp proj; in-place res)
// MODE 3: f32 out = acc                  (logits)
// ---------------------------------------------------------------------------
template <int MODE>
__global__ __launch_bounds__(256)
void gemm_wmma_kernel(const _Float16* __restrict__ A,
                      const _Float16* __restrict__ Bt,
                      const float* __restrict__ bias,
                      const float* resid, float* Cf,
                      _Float16* __restrict__ Ch,
                      int M, int N, int K) {
  const int wave = threadIdx.x >> 5;
  const int lane = threadIdx.x & 31;
  const int wm = wave >> 1;        // 0..3
  const int wn = wave & 1;         // 0..1
  const int m0 = blockIdx.y * 128 + wm * 32;
  const int n0 = blockIdx.x * 64  + wn * 32;

  v8f acc00 = zero8(), acc01 = zero8(), acc10 = zero8(), acc11 = zero8();

  const _Float16* Ar0 = A  + (size_t)m0 * K;
  const _Float16* Ar1 = A  + (size_t)(m0 + 16) * K;
  const _Float16* Br0 = Bt + (size_t)n0 * K;
  const _Float16* Br1 = Bt + (size_t)(n0 + 16) * K;

  // prologue: fragments for k0 = 0
  v16h a0 = load_frag(Ar0, K);
  v16h a1 = load_frag(Ar1, K);
  v16h b0 = load_frag(Br0, K);
  v16h b1 = load_frag(Br1, K);

  for (int k0 = 32; k0 < K; k0 += 32) {
    __builtin_prefetch(Ar0 + k0 + 64, 0, 1);
    __builtin_prefetch(Br0 + k0 + 64, 0, 1);
    v16h na0 = load_frag(Ar0 + k0, K);
    v16h na1 = load_frag(Ar1 + k0, K);
    v16h nb0 = load_frag(Br0 + k0, K);
    v16h nb1 = load_frag(Br1 + k0, K);
    acc00 = wmma32(a0, b0, acc00);
    acc01 = wmma32(a0, b1, acc01);
    acc10 = wmma32(a1, b0, acc10);
    acc11 = wmma32(a1, b1, acc11);
    a0 = na0; a1 = na1; b0 = nb0; b1 = nb1;
  }
  acc00 = wmma32(a0, b0, acc00);
  acc01 = wmma32(a0, b1, acc01);
  acc10 = wmma32(a1, b0, acc10);
  acc11 = wmma32(a1, b1, acc11);

  const int nl = lane & 15;
  const int mh = (lane >> 4) << 3;
#pragma unroll
  for (int fm = 0; fm < 2; ++fm) {
#pragma unroll
    for (int fn = 0; fn < 2; ++fn) {
      v8f acc = (fm == 0) ? (fn == 0 ? acc00 : acc01) : (fn == 0 ? acc10 : acc11);
      int gn = n0 + fn * 16 + nl;
      float bi = 0.0f;
      if constexpr (MODE != 3) bi = bias[gn];
#pragma unroll
      for (int i = 0; i < 8; ++i) {
        int gm = m0 + fm * 16 + mh + i;
        float v = acc[i] + bi;
        size_t o = (size_t)gm * N + gn;
        if constexpr (MODE == 0) {
          Ch[o] = (_Float16)v;
        } else if constexpr (MODE == 1) {
          Ch[o] = (_Float16)(v / (1.0f + __expf(-v)));   // silu
        } else if constexpr (MODE == 2) {
          Cf[o] = v + resid[o];
        } else {
          Cf[o] = v;
        }
      }
    }
  }
}

// ---------------------------------------------------------------------------
// Flash attention, causal GQA (NH=9 q-heads, NKV=3 kv-heads, HD=64, T=1024).
// Grid: (T/64, B*NH). 128 threads = 4 waves; wave handles 16 query rows.
// Per 32-key block: K and V tiles DMA'd to LDS by the TDM (wave 0 issues
// tensor_load_to_lds, waits s_wait_tensorcnt), V transposed in LDS for the
// Bt fragment of P@V. S = Q K^T and O += P V both on WMMA; online softmax.
// All fragments for each WMMA group are loaded before the WMMAs issue so the
// ds_load_b128s batch under one s_wait_dscnt.
// ---------------------------------------------------------------------------
__global__ __launch_bounds__(128)
void attn_kernel(const _Float16* __restrict__ q,
                 const _Float16* __restrict__ kbuf,
                 const _Float16* __restrict__ vbuf,
                 _Float16* __restrict__ y) {
  const int bh = blockIdx.y;
  const int b  = bh / C_NH;
  const int h  = bh - b * C_NH;
  const int g  = h / 3;                 // kv head = h // N_REP
  const int qblock = blockIdx.x;        // 64-query block

  const int wave = threadIdx.x >> 5;
  const int lane = threadIdx.x & 31;
  const int nl = lane & 15;
  const int mh = (lane >> 4) << 3;
  const int q0 = qblock * 64 + wave * 16;

  const _Float16* qhead = q    + ((size_t)(b * C_T + q0)) * C_D + h * C_HD;
  const _Float16* khead = kbuf + ((size_t)b * C_T) * C_KVW + g * C_HD;
  const _Float16* vhead = vbuf + ((size_t)b * C_T) * C_KVW + g * C_HD;

  // Q fragments for this wave's 16 rows (d = 0..31, 32..63)
  v16h aq0 = load_frag(qhead,      C_D);
  v16h aq1 = load_frag(qhead + 32, C_D);

  __shared__ alignas(16) _Float16 Pl[4][16 * 32];   // per-wave P tile
  __shared__ alignas(16) _Float16 Kst[32 * 64];     // K tile, row-major
  __shared__ alignas(16) _Float16 Vst[32 * 64];     // V tile, row-major
  __shared__ alignas(16) _Float16 Vt[64 * 40];      // V^T tile [d][k], ld=40
  _Float16* Pw = Pl[wave];

  float mrun[8], lrun[8];
  v8f Of0 = zero8(), Of1 = zero8(), Of2 = zero8(), Of3 = zero8();
#pragma unroll
  for (int i = 0; i < 8; ++i) { mrun[i] = -1e30f; lrun[i] = 0.0f; }

  const int nkb = qblock * 2 + 2;       // 32-key blocks up to & incl. diagonal
  for (int kb = 0; kb < nkb; ++kb) {
    __syncthreads();                    // prior-iteration consumers done
#if HAS_TDM
    if (threadIdx.x < 32) {             // one wave drives the Tensor Data Mover
      tdm_load_tile32x64((unsigned)(uintptr_t)Kst,
                         khead + (size_t)(kb * 32) * C_KVW, C_KVW);
      tdm_load_tile32x64((unsigned)(uintptr_t)Vst,
                         vhead + (size_t)(kb * 32) * C_KVW, C_KVW);
      __builtin_amdgcn_s_wait_tensorcnt(0);
    }
    __syncthreads();
#else
    for (int idx = threadIdx.x; idx < 32 * 64; idx += 128) {
      int kk = idx >> 6, d = idx & 63;
      Kst[kk * 64 + d] = khead[(size_t)(kb * 32 + kk) * C_KVW + d];
      Vst[kk * 64 + d] = vhead[(size_t)(kb * 32 + kk) * C_KVW + d];
    }
    __syncthreads();
#endif
    // transpose V tile in LDS for the Bt fragment of P@V
    for (int idx = threadIdx.x; idx < 32 * 64; idx += 128) {
      int kk = idx >> 6, d = idx & 63;
      Vt[d * 40 + kk] = Vst[kk * 64 + d];
    }
    __syncthreads();

    // load all four K fragments first, then issue the four WMMAs
    v16h kf00 = load_frag(Kst,                64);
    v16h kf01 = load_frag(Kst + 32,           64);
    v16h kf10 = load_frag(Kst + 16 * 64,      64);
    v16h kf11 = load_frag(Kst + 16 * 64 + 32, 64);
    v8f s0 = zero8(), s1 = zero8();
    s0 = wmma32(aq0, kf00, s0);
    s1 = wmma32(aq0, kf10, s1);
    s0 = wmma32(aq1, kf01, s0);
    s1 = wmma32(aq1, kf11, s1);

    // online softmax per row (rows live per-register across 16 lanes)
#pragma unroll
    for (int i = 0; i < 8; ++i) {
      int qi = q0 + mh + i;
      float x0 = (kb * 32 + nl      <= qi) ? s0[i] * 0.125f : -1e30f;
      float x1 = (kb * 32 + 16 + nl <= qi) ? s1[i] * 0.125f : -1e30f;
      float tmax = fmaxf(x0, x1);
#pragma unroll
      for (int o = 8; o >= 1; o >>= 1) tmax = fmaxf(tmax, __shfl_xor(tmax, o, 32));
      float mnew = fmaxf(mrun[i], tmax);
      float fac  = __expf(mrun[i] - mnew);
      float p0 = __expf(x0 - mnew);
      float p1 = __expf(x1 - mnew);
      float rs = p0 + p1;
#pragma unroll
      for (int o = 8; o >= 1; o >>= 1) rs += __shfl_xor(rs, o, 32);
      lrun[i] = lrun[i] * fac + rs;
      mrun[i] = mnew;
      Pw[(mh + i) * 32 + nl]      = (_Float16)p0;
      Pw[(mh + i) * 32 + 16 + nl] = (_Float16)p1;
      Of0[i] *= fac; Of1[i] *= fac; Of2[i] *= fac; Of3[i] *= fac;
    }

    // O += P @ V : load P + all V^T fragments, then issue the four WMMAs
    v16h ap  = load_frag(Pw, 32);
    v16h vf0 = load_frag(Vt,           40);
    v16h vf1 = load_frag(Vt + 16 * 40, 40);
    v16h vf2 = load_frag(Vt + 32 * 40, 40);
    v16h vf3 = load_frag(Vt + 48 * 40, 40);
    Of0 = wmma32(ap, vf0, Of0);
    Of1 = wmma32(ap, vf1, Of1);
    Of2 = wmma32(ap, vf2, Of2);
    Of3 = wmma32(ap, vf3, Of3);
  }

  _Float16* yhead = y + ((size_t)(b * C_T + q0)) * C_D + h * C_HD;
#pragma unroll
  for (int i = 0; i < 8; ++i) {
    float inv = 1.0f / lrun[i];
    size_t ro = (size_t)(mh + i) * C_D;
    yhead[ro +      nl] = (_Float16)(Of0[i] * inv);
    yhead[ro + 16 + nl] = (_Float16)(Of1[i] * inv);
    yhead[ro + 32 + nl] = (_Float16)(Of2[i] * inv);
    yhead[ro + 48 + nl] = (_Float16)(Of3[i] * inv);
  }
}

// ---------------------------------------------------------------------------
// loss: per row logsumexp over V, nll gather, mean via atomicAdd
// ---------------------------------------------------------------------------
__global__ void zero1_kernel(float* p) { if (threadIdx.x == 0) *p = 0.0f; }

__global__ __launch_bounds__(256)
void loss_kernel(const float* __restrict__ logits, const int* __restrict__ targets,
                 float* __restrict__ loss, float invBT) {
  const int row  = blockIdx.x;
  const int lane = threadIdx.x & 31;
  const int wv   = threadIdx.x >> 5;
  const float* lr = logits + (size_t)row * C_V;

  __shared__ float red[8];
  __shared__ float sval;

  float mx = -1e30f;
  for (int i = threadIdx.x; i < C_V; i += 256) mx = fmaxf(mx, lr[i]);
#pragma unroll
  for (int o = 16; o >= 1; o >>= 1) mx = fmaxf(mx, __shfl_xor(mx, o, 32));
  if (lane == 0) red[wv] = mx;
  __syncthreads();
  if (threadIdx.x == 0) {
    float m = red[0];
#pragma unroll
    for (int j = 1; j < 8; ++j) m = fmaxf(m, red[j]);
    sval = m;
  }
  __syncthreads();
  mx = sval;

  float se = 0.0f;
  for (int i = threadIdx.x; i < C_V; i += 256) se += __expf(lr[i] - mx);
#pragma unroll
  for (int o = 16; o >= 1; o >>= 1) se += __shfl_xor(se, o, 32);
  if (lane == 0) red[wv] = se;
  __syncthreads();
  if (threadIdx.x == 0) {
    float s = 0.0f;
#pragma unroll
    for (int j = 0; j < 8; ++j) s += red[j];
    float lse = mx + __logf(s);
    float nll = lse - lr[targets[row]];
    atomicAdd(loss, nll * invBT);
  }
}

// ---------------------------------------------------------------------------
// host orchestration
// ---------------------------------------------------------------------------
extern "C" void kernel_launch(void* const* d_in, const int* in_sizes, int n_in,
                              void* d_out, int out_size, void* d_ws, size_t ws_size,
                              hipStream_t stream) {
  // setup_inputs() dict order (params first, insertion order, then idx/targets)
  const float* wte    = (const float*)d_in[0];
  const float* wpe    = (const float*)d_in[1];
  const float* ln1_w  = (const float*)d_in[2];
  const float* ln1_b  = (const float*)d_in[3];
  const float* wq     = (const float*)d_in[4];
  const float* bq     = (const float*)d_in[5];
  const float* wk     = (const float*)d_in[6];
  const float* bk     = (const float*)d_in[7];
  const float* wv     = (const float*)d_in[8];
  const float* bv     = (const float*)d_in[9];
  const float* wo     = (const float*)d_in[10];
  const float* bo     = (const float*)d_in[11];
  const float* ln2_w  = (const float*)d_in[12];
  const float* ln2_b  = (const float*)d_in[13];
  const float* wfc    = (const float*)d_in[14];
  const float* bfc    = (const float*)d_in[15];
  const float* wproj  = (const float*)d_in[16];
  const float* bproj  = (const float*)d_in[17];
  const float* lnf_w  = (const float*)d_in[18];
  const float* lnf_b  = (const float*)d_in[19];
  const float* lmh    = (const float*)d_in[20];
  const int*   idx    = (const int*)d_in[21];
  const int*   tgt    = (const int*)d_in[22];

  float* out = (float*)d_out;                 // logits [M,V] then loss [1]
  float* loss_ptr = out + (size_t)C_M * C_V;

  // bump allocator over workspace
  char* ws = (char*)d_ws;
  size_t off = 0;
  auto alloc = [&](size_t bytes) -> char* {
    char* p = ws + off;
    off = (off + bytes + 255) & ~(size_t)255;
    return p;
  };

  // f16 transposed weights [N,K]
  _Float16* wqT   = (_Float16*)alloc((size_t)C_NL * C_D * C_D   * 2);
  _Float16* wkT   = (_Float16*)alloc((size_t)C_NL * C_KVW * C_D * 2);
  _Float16* wvT   = (_Float16*)alloc((size_t)C_NL * C_KVW * C_D * 2);
  _Float16* woT   = (_Float16*)alloc((size_t)C_NL * C_D * C_D   * 2);
  _Float16* wfcT  = (_Float16*)alloc((size_t)C_NL * C_MLP * C_D * 2);
  _Float16* wprT  = (_Float16*)alloc((size_t)C_NL * C_D * C_MLP * 2);
  _Float16* lmh16 = (_Float16*)alloc((size_t)C_V * C_D * 2);     // already [N,K]
  // activations
  float*    x    = (float*)   alloc((size_t)C_M * C_D * 4);
  _Float16* h16  = (_Float16*)alloc((size_t)C_M * C_D * 2);
  _Float16* q16  = (_Float16*)alloc((size_t)C_M * C_D * 2);
  _Float16* k16  = (_Float16*)alloc((size_t)C_M * C_KVW * 2);
  _Float16* v16  = (_Float16*)alloc((size_t)C_M * C_KVW * 2);
  _Float16* y16  = (_Float16*)alloc((size_t)C_M * C_D * 2);
  _Float16* fc16 = (_Float16*)alloc((size_t)C_M * C_MLP * 2);
  _Float16* xf16 = (_Float16*)alloc((size_t)C_M * C_D * 2);

  // ---- weight conversion (deterministic every call) ----
  auto tgrid = [](int n) { return (n + 255) / 256; };
  for (int l = 0; l < C_NL; ++l) {
    transpose_f16_kernel<<<tgrid(C_D * C_D),   256, 0, stream>>>(wq    + (size_t)l * C_D * C_D,   wqT  + (size_t)l * C_D * C_D,   C_D,   C_D);
    transpose_f16_kernel<<<tgrid(C_D * C_KVW), 256, 0, stream>>>(wk    + (size_t)l * C_D * C_KVW, wkT  + (size_t)l * C_KVW * C_D, C_D,   C_KVW);
    transpose_f16_kernel<<<tgrid(C_D * C_KVW), 256, 0, stream>>>(wv    + (size_t)l * C_D * C_KVW, wvT  + (size_t)l * C_KVW * C_D, C_D,   C_KVW);
    transpose_f16_kernel<<<tgrid(C_D * C_D),   256, 0, stream>>>(wo    + (size_t)l * C_D * C_D,   woT  + (size_t)l * C_D * C_D,   C_D,   C_D);
    transpose_f16_kernel<<<tgrid(C_D * C_MLP), 256, 0, stream>>>(wfc   + (size_t)l * C_D * C_MLP, wfcT + (size_t)l * C_MLP * C_D, C_D,   C_MLP);
    transpose_f16_kernel<<<tgrid(C_MLP * C_D), 256, 0, stream>>>(wproj + (size_t)l * C_MLP * C_D, wprT + (size_t)l * C_D * C_MLP, C_MLP, C_D);
  }
  cast_f16_kernel<<<4096, 256, 0, stream>>>(lmh, lmh16, (long long)C_V * C_D);

  // ---- embeddings ----
  embed_kernel<<<C_M, 256, 0, stream>>>(idx, wte, wpe, x);

  const dim3 gemm_block(256);
  auto gemm_grid = [](int N) { return dim3((unsigned)(N / 64), (unsigned)(C_M / 128)); };

  // ---- transformer layers ----
  for (int l = 0; l < C_NL; ++l) {
    layernorm_f16_kernel<<<C_M, 256, 0, stream>>>(x, ln1_w + (size_t)l * C_D, ln1_b + (size_t)l * C_D, h16);

    gemm_wmma_kernel<0><<<gemm_grid(C_D), gemm_block, 0, stream>>>(
        h16, wqT + (size_t)l * C_D * C_D, bq + (size_t)l * C_D,
        nullptr, nullptr, q16, C_M, C_D, C_D);
    gemm_wmma_kernel<0><<<gemm_grid(C_KVW), gemm_block, 0, stream>>>(
        h16, wkT + (size_t)l * C_KVW * C_D, bk + (size_t)l * C_KVW,
        nullptr, nullptr, k16, C_M, C_KVW, C_D);
    gemm_wmma_kernel<0><<<gemm_grid(C_KVW), gemm_block, 0, stream>>>(
        h16, wvT + (size_t)l * C_KVW * C_D, bv + (size_t)l * C_KVW,
        nullptr, nullptr, v16, C_M, C_KVW, C_D);

    attn_kernel<<<dim3(C_T / 64, C_B * C_NH), 128, 0, stream>>>(q16, k16, v16, y16);

    gemm_wmma_kernel<2><<<gemm_grid(C_D), gemm_block, 0, stream>>>(
        y16, woT + (size_t)l * C_D * C_D, bo + (size_t)l * C_D,
        x, x, nullptr, C_M, C_D, C_D);

    layernorm_f16_kernel<<<C_M, 256, 0, stream>>>(x, ln2_w + (size_t)l * C_D, ln2_b + (size_t)l * C_D, h16);

    gemm_wmma_kernel<1><<<gemm_grid(C_MLP), gemm_block, 0, stream>>>(
        h16, wfcT + (size_t)l * C_MLP * C_D, bfc + (size_t)l * C_MLP,
        nullptr, nullptr, fc16, C_M, C_MLP, C_D);
    gemm_wmma_kernel<2><<<gemm_grid(C_D), gemm_block, 0, stream>>>(
        fc16, wprT + (size_t)l * C_D * C_MLP, bproj + (size_t)l * C_D,
        x, x, nullptr, C_M, C_D, C_MLP);
  }

  // ---- final LN + logits + loss ----
  layernorm_f16_kernel<<<C_M, 256, 0, stream>>>(x, lnf_w, lnf_b, xf16);
  gemm_wmma_kernel<3><<<gemm_grid(C_V), gemm_block, 0, stream>>>(
      xf16, lmh16, nullptr, nullptr, out, nullptr, C_M, C_V, C_D);

  zero1_kernel<<<1, 32, 0, stream>>>(loss_ptr);
  loss_kernel<<<C_M, 256, 0, stream>>>(out, tgt, loss_ptr, 1.0f / (float)C_M);
}
